// SinkhornCausalAttention_71597104824510
// MI455X (gfx1250) — matrix operand order repaired and checked by reading, not compiled, not run on hardware
//
#include <hip/hip_runtime.h>
#include <hip/hip_bf16.h>

// ---------------- constants ----------------
constexpr int TT   = 4096;  // sequence length
constexpr int DHC  = 128;   // head dim
constexpr int NBUK = 32;    // buckets
constexpr int BSZ  = 128;   // bucket size
constexpr int HC   = 8;     // heads
constexpr int HH   = 4;     // rotated-head split

constexpr int K2_STRIDE = 136; // halves per row (272B: 16B-aligned frags, bank step 4)
constexpr int VT_STRIDE = 264; // halves per row (528B)
constexpr int P_STRIDE  = 264;

constexpr float NEGMAX = -3.4028235e38f;

typedef __attribute__((ext_vector_type(16))) __bf16 v16bf;
typedef __attribute__((ext_vector_type(8)))  float  v8f;

union FragU { uint4 u[2]; v16bf bf; };

// f32 -> bf16 (round to nearest even)
__device__ __forceinline__ unsigned short f2bf(float f) {
  union { float f; unsigned u; } c{f};
  unsigned r = c.u + 0x7FFFu + ((c.u >> 16) & 1u);
  return (unsigned short)(r >> 16);
}

// A-operand frag (16x32 bf16): lane L holds row M=L%16;
// lanes<16: K in {0..7}U{16..23};  lanes>=16: K in {8..15}U{24..31}
__device__ __forceinline__ v16bf ld_fragA_lds(const unsigned short* row, int kbase, int half) {
  FragU fu;
  fu.u[0] = *(const uint4*)(row + kbase + half * 8);
  fu.u[1] = *(const uint4*)(row + kbase + 16 + half * 8);
  return fu.bf;
}

// B-operand frag (32x16 bf16): lane L holds col N=L%16;
// lanes<16: K in {0..7}U{16..23};  lanes>=16: K in {8..15}U{24..31}
__device__ __forceinline__ v16bf ld_fragB_lds(const unsigned short* row, int kbase, int half) {
  FragU fu;
  fu.u[0] = *(const uint4*)(row + kbase + half * 8);
  fu.u[1] = *(const uint4*)(row + kbase + 16 + half * 8);
  return fu.bf;
}

// A-operand frag straight from global f32 (scaled, converted to bf16)
__device__ __forceinline__ v16bf ld_fragA_f32(const float* row, int kbase, int half, float sc) {
  v16bf r;
#pragma unroll
  for (int t = 0; t < 8; ++t) r[t]     = (__bf16)(row[kbase + half * 8 + t] * sc);
#pragma unroll
  for (int t = 0; t < 8; ++t) r[8 + t] = (__bf16)(row[kbase + 16 + half * 8 + t] * sc);
  return r;
}

// rotated-time mapping: heads >= HH use roll(x,-127) => load index (t+127) mod T
__device__ __forceinline__ int tmap(int tr, bool rot) {
  return rot ? ((tr + 127) & (TT - 1)) : tr;
}

// ============ Kernel A1: per-bucket sums of k + first element of each bucket ============
__global__ void kbucket_sums(const float* __restrict__ k, float* __restrict__ Sb,
                             float* __restrict__ X) {
  int bid = blockIdx.x;             // bh*32 + u
  int u = bid & 31;
  int bh = bid >> 5;
  int h = bh & 7;
  bool rot = h >= HH;
  int d = threadIdx.x;              // 0..127
  const float* kb = k + (size_t)bh * TT * DHC;
  float acc = 0.f;
  for (int it = 0; it < BSZ; ++it) {
    int tg = tmap(u * BSZ + it, rot);
    float kv = kb[(size_t)tg * DHC + d];
    acc += kv;
    if (it == 0) X[((size_t)bh * NBUK + u) * 256 + 128 + d] = kv; // x second half: k at bucket start
  }
  Sb[((size_t)bh * NBUK + u) * DHC + d] = acc;
}

// ============ Kernel A2: prefix over buckets -> running mean at bucket starts ============
__global__ void kbucket_prefix(const float* __restrict__ Sb, float* __restrict__ X) {
  int bh = blockIdx.x;
  int d = threadIdx.x;
  float acc = 0.f;
  for (int u = 0; u < NBUK; ++u) {
    float kf = X[((size_t)bh * NBUK + u) * 256 + 128 + d];
    X[((size_t)bh * NBUK + u) * 256 + d] = (acc + kf) / (float)(u * BSZ + 1);
    acc += Sb[((size_t)bh * NBUK + u) * DHC + d];
  }
}

// ============ Kernel B: routing R = tril(softmax(leaky_relu(x @ W)),-1); keep row max ============
__global__ void routing(const float* __restrict__ X, const float* __restrict__ W,
                        float* __restrict__ rval, int* __restrict__ ridx) {
  int bh = blockIdx.x;
  int h = bh & 7;
  __shared__ float M[32][33];
  const float* Xb = X + (size_t)bh * NBUK * 256;
  const float* Wh = W + (size_t)h * 256 * NBUK;   // W_sort[0,h,:,:] (256 x 32)
  int tid = threadIdx.x;                          // 256 threads
#pragma unroll
  for (int e2 = 0; e2 < 4; ++e2) {
    int ent = tid * 4 + e2;
    int uu = ent >> 5;
    int vv = ent & 31;
    float acc = 0.f;
    for (int e = 0; e < 256; ++e) acc += Xb[uu * 256 + e] * Wh[e * 32 + vv];
    M[uu][vv] = (acc >= 0.f) ? acc : 0.01f * acc; // leaky_relu
  }
  __syncthreads();
  if (tid < 32) {
    int uu = tid;
    float mx = NEGMAX;
    for (int vv = 0; vv < 32; ++vv) mx = fmaxf(mx, M[uu][vv]);
    float sm = 0.f;
    for (int vv = 0; vv < 32; ++vv) sm += __expf(M[uu][vv] - mx);
    float inv = 1.f / sm;
    float best = -1.f;
    int bi = 0;
    for (int vv = 0; vv < uu; ++vv) {       // tril(.,-1): only v<u survive
      float pv = __expf(M[uu][vv] - mx) * inv;
      if (pv > best) { best = pv; bi = vv; }
    }
    if (uu == 0) { best = 0.f; bi = 0; }    // row 0 fully zeroed by tril
    rval[(size_t)bh * NBUK + uu] = best;
    ridx[(size_t)bh * NBUK + uu] = bi;
  }
}

// ============ Kernel C: per-bucket attention with WMMA bf16 ============
__global__ __launch_bounds__(256) void attn_wmma(
    const float* __restrict__ q, const float* __restrict__ k, const float* __restrict__ v,
    const float* __restrict__ rval, const int* __restrict__ ridx, float* __restrict__ out) {
  // LDS: K2 (256x136 halves) + V2T (128x264 halves); P strip aliases K2 after barrier.
  __shared__ __align__(16) char smem[256 * K2_STRIDE * 2 + 128 * VT_STRIDE * 2]; // 137216 B
  unsigned short* K2s = (unsigned short*)smem;                       // 69632 B
  unsigned short* V2s = (unsigned short*)(smem + 256 * K2_STRIDE * 2);
  unsigned short* Ps  = (unsigned short*)smem;                       // 128*264*2 = 67584 B (alias)

  int bid = blockIdx.x;          // b*H*NBUK + h*NBUK + u
  int u = bid & 31;
  int h = (bid >> 5) & 7;
  int b = bid >> 8;
  bool rot = h >= HH;
  bool lastrot = rot && (u == NBUK - 1);
  size_t bh = (size_t)(b * HC + h);

  const float* qb = q + bh * TT * DHC;
  const float* kb = k + bh * TT * DHC;
  const float* vb = v + bh * TT * DHC;

  int tid = threadIdx.x;
  float sval = rval[bh * NBUK + u];
  int   sidx = ridx[bh * NBUK + u];

  // ---- stage K2 (gathered-scaled ++ current) and V2^T into LDS as bf16 ----
  int tg = sidx * BSZ;
  int tc = u * BSZ;
  for (int idx = tid; idx < 256 * DHC; idx += 256) {
    int j = idx >> 7;        // 0..255 key index
    int d = idx & 127;       // dh
    int t_rs = (j < BSZ) ? (tg + j) : (tc + j - BSZ);
    int t_g = tmap(t_rs, rot);
    float sc = (j < BSZ) ? sval : 1.0f;
    float kv = kb[(size_t)t_g * DHC + d] * sc;
    float vv = vb[(size_t)t_g * DHC + d] * sc;
    K2s[j * K2_STRIDE + d] = f2bf(kv);
    V2s[d * VT_STRIDE + j] = f2bf(vv);
  }
  __syncthreads();

  int lane = tid & 31;
  int w = tid >> 5;          // wave id: rows w*16..w*16+15
  int half = lane >> 4;      // 0 or 1
  int nloc = lane & 15;

  // ---- Q A-fragments straight from global (rotated row map), pre-scaled by d^-0.5 ----
  int qrow_g = tmap(u * BSZ + w * 16 + nloc, rot);
  const float* qrow = qb + (size_t)qrow_g * DHC;
  v16bf aq[4];
#pragma unroll
  for (int kt = 0; kt < 4; ++kt) aq[kt] = ld_fragA_f32(qrow, kt * 32, half, 0.03125f);

  // ---- S = Q @ K2^T : 16 column tiles, K=128 in 4 steps; double-buffered B frags ----
  v8f S[16];
  v16bf b0[4], b1[4];
  {
    const unsigned short* kr = K2s + nloc * K2_STRIDE;
#pragma unroll
    for (int kt = 0; kt < 4; ++kt) b0[kt] = ld_fragB_lds(kr, kt * 32, half);
  }
#pragma unroll
  for (int nt = 0; nt < 16; nt += 2) {
    {   // prefetch tile nt+1 into b1 while WMMA consumes b0
      const unsigned short* kr = K2s + ((nt + 1) * 16 + nloc) * K2_STRIDE;
#pragma unroll
      for (int kt = 0; kt < 4; ++kt) b1[kt] = ld_fragB_lds(kr, kt * 32, half);
    }
    {
      v8f acc = {};
#pragma unroll
      for (int kt = 0; kt < 4; ++kt)
        acc = __builtin_amdgcn_wmma_f32_16x16x32_bf16(false, aq[kt], false, b0[kt],
                                                      (short)0, acc, false, false);
      S[nt] = acc;
    }
    if (nt + 2 < 16) {  // prefetch tile nt+2 into b0 while WMMA consumes b1
      const unsigned short* kr = K2s + ((nt + 2) * 16 + nloc) * K2_STRIDE;
#pragma unroll
      for (int kt = 0; kt < 4; ++kt) b0[kt] = ld_fragB_lds(kr, kt * 32, half);
    }
    {
      v8f acc = {};
#pragma unroll
      for (int kt = 0; kt < 4; ++kt)
        acc = __builtin_amdgcn_wmma_f32_16x16x32_bf16(false, aq[kt], false, b1[kt],
                                                      (short)0, acc, false, false);
      S[nt + 1] = acc;
    }
  }
  __syncthreads();   // all waves finished reading K2s -> safe to alias Ps over it

  // ---- mask + row softmax (rows live within one 16-lane half); S already scaled via Q ----
#pragma unroll
  for (int g = 0; g < 8; ++g) {
    int i_loc = w * 16 + half * 8 + g;          // query row in bucket (0..127)
    float rmax = NEGMAX;
#pragma unroll
    for (int nt = 0; nt < 16; ++nt) {
      int j = nt * 16 + nloc;
      float s = S[nt][g];
      bool ok = (j < BSZ) || (i_loc >= j - BSZ);
      if (lastrot && (j <= BSZ) && (i_loc > 0)) ok = false;
      s = ok ? s : NEGMAX;
      S[nt][g] = s;
      rmax = fmaxf(rmax, s);
    }
#pragma unroll
    for (int off = 1; off < 16; off <<= 1) rmax = fmaxf(rmax, __shfl_xor(rmax, off, 32));
    float rsum = 0.f;
#pragma unroll
    for (int nt = 0; nt < 16; ++nt) {
      float p = __expf(S[nt][g] - rmax);
      S[nt][g] = p;
      rsum += p;
    }
#pragma unroll
    for (int off = 1; off < 16; off <<= 1) rsum += __shfl_xor(rsum, off, 32);
    float inv = 1.f / rsum;
    unsigned short* pr = Ps + (w * 16 + half * 8 + g) * P_STRIDE;
#pragma unroll
    for (int nt = 0; nt < 16; ++nt) pr[nt * 16 + nloc] = f2bf(S[nt][g] * inv);
  }

  // wave-private P: order our own LDS stores before re-reading (cross-lane)
  asm volatile("s_wait_dscnt 0" ::: "memory");

  // ---- O = P @ V2 : A from wave's P strip, B from V2^T; double-buffered half-K tiles ----
  v16bf ap[8];
  const unsigned short* prow = Ps + (w * 16 + nloc) * P_STRIDE;
#pragma unroll
  for (int kt = 0; kt < 8; ++kt) ap[kt] = ld_fragA_lds(prow, kt * 32, half);

  float* ob = out + bh * TT * DHC;
  v16bf c0[4], c1[4];
  {   // preload first half-K tile of nt=0
    const unsigned short* vr = V2s + nloc * VT_STRIDE;
#pragma unroll
    for (int kt = 0; kt < 4; ++kt) c0[kt] = ld_fragB_lds(vr, kt * 32, half);
  }
#pragma unroll
  for (int nt = 0; nt < 8; ++nt) {
    const unsigned short* vr = V2s + (nt * 16 + nloc) * VT_STRIDE;
    {   // prefetch second half-K of this tile into c1
#pragma unroll
      for (int kt = 0; kt < 4; ++kt) c1[kt] = ld_fragB_lds(vr, (4 + kt) * 32, half);
    }
    v8f acc = {};
#pragma unroll
    for (int kt = 0; kt < 4; ++kt)
      acc = __builtin_amdgcn_wmma_f32_16x16x32_bf16(false, ap[kt], false, c0[kt],
                                                    (short)0, acc, false, false);
    if (nt < 7) {  // prefetch first half-K of next tile into c0
      const unsigned short* vr2 = V2s + ((nt + 1) * 16 + nloc) * VT_STRIDE;
#pragma unroll
      for (int kt = 0; kt < 4; ++kt) c0[kt] = ld_fragB_lds(vr2, kt * 32, half);
    }
#pragma unroll
    for (int kt = 0; kt < 4; ++kt)
      acc = __builtin_amdgcn_wmma_f32_16x16x32_bf16(false, ap[4 + kt], false, c1[kt],
                                                    (short)0, acc, false, false);
#pragma unroll
    for (int g = 0; g < 8; ++g) {
      int i_loc = w * 16 + half * 8 + g;
      int t_g = tmap(u * BSZ + i_loc, rot);     // inverse rotation == same map
      ob[(size_t)t_g * DHC + nt * 16 + nloc] = acc[g];
    }
  }
}

// ---------------- host launcher ----------------
extern "C" void kernel_launch(void* const* d_in, const int* in_sizes, int n_in,
                              void* d_out, int out_size, void* d_ws, size_t ws_size,
                              hipStream_t stream) {
  const float* q = (const float*)d_in[0];
  const float* k = (const float*)d_in[1];
  const float* v = (const float*)d_in[2];
  const float* W = (const float*)d_in[3];
  float* out = (float*)d_out;

  char* ws = (char*)d_ws;
  float* Sb   = (float*)ws;                  // 4*8*32*128*4 = 524288 B
  float* X    = (float*)(ws + 524288);       // 4*8*32*256*4 = 1048576 B
  float* rv   = (float*)(ws + 1572864);      // 4096 B
  int*   ri   = (int*)  (ws + 1576960);      // 4096 B

  kbucket_sums  <<<dim3(32 * NBUK), dim3(128), 0, stream>>>(k, Sb, X);
  kbucket_prefix<<<dim3(32),        dim3(128), 0, stream>>>(Sb, X);
  routing       <<<dim3(32),        dim3(256), 0, stream>>>(X, W, rv, ri);
  attn_wmma     <<<dim3(4 * HC * NBUK), dim3(256), 0, stream>>>(q, k, v, rv, ri, out);
}
